// RNN_67997922230401
// MI455X (gfx1250) — matrix operand order repaired and checked by reading, not compiled, not run on hardware
//
#include <hip/hip_runtime.h>
#include <hip/hip_bf16.h>
#include <stdint.h>

// ---------------------------------------------------------------------------
// Problem dims
// ---------------------------------------------------------------------------
#define B_DIM 128
#define S_DIM 512
#define C_DIM 1104
#define H_DIM 70
#define G_DIM 280          // 4*H
#define GP    288          // gates padded to 18*16
#define KCH   35           // ceil(1104/32) -> padded K = 1120
#define NROWS (B_DIM * S_DIM)
#define OSTR  72           // padded hidden stride for LSTM outputs
#define BCHUNK 4608        // dwords per Wc K-chunk slab (18*2*16*8 == 16*288)

typedef __attribute__((ext_vector_type(16))) __bf16 v16bf;
typedef __attribute__((ext_vector_type(8)))  float  v8f;

union FragH { unsigned u[8]; v16bf v; };
union FragF { float    f[8]; v8f   v; };

static __device__ __forceinline__ unsigned f2bf(float x) {
    unsigned u = __float_as_uint(x);
    return (u + 0x7FFFu + ((u >> 16) & 1u)) >> 16;   // RNE
}
static __device__ __forceinline__ unsigned pack2(float a, float b) {
    return f2bf(a) | (f2bf(b) << 16);
}
// low 32 bits of a flat LDS pointer == LDS byte address (ISA 10.2 aperture rule)
static __device__ __forceinline__ unsigned ldsAddr(const void* p) {
    return (unsigned)(uintptr_t)p;
}
static __device__ __forceinline__ void asyncCopyB128(unsigned lds, unsigned long long ga) {
    asm volatile("global_load_async_to_lds_b128 %0, %1, off" :: "v"(lds), "v"(ga) : "memory");
}
static __device__ __forceinline__ void asyncWait0() {
    asm volatile("s_wait_asynccnt 0" ::: "memory");
}

// ---------------------------------------------------------------------------
// K0: Wc = renorm(embed_w) @ w_ih^T, stored FRAGMENT-MAJOR per K-chunk:
//     wcP[ck][tile(18)][half(2)][col(16)][j(8)]  (bf16 K-pair dwords)
//     so a lane's 8 fragment dwords are contiguous in LDS after async copy.
//     Also w_hh^T pair-packed row-major (48x288), bias = b_ih+b_hh (288, pad 0)
// ---------------------------------------------------------------------------
__global__ __launch_bounds__(256)
void prep_kernel(const float* __restrict__ ew,  const float* __restrict__ wih,
                 const float* __restrict__ whh, const float* __restrict__ bih,
                 const float* __restrict__ bhh,
                 unsigned* __restrict__ wcP, unsigned* __restrict__ whhP,
                 float* __restrict__ bias) {
    const int WC_N  = KCH * BCHUNK;      // 161280
    const int WHH_N = 48 * GP;           // 13824
    int idx = blockIdx.x * 256 + threadIdx.x;
    if (idx < WC_N) {
        int j  = idx & 7;
        int c  = (idx >> 3) & 15;
        int hb = (idx >> 7) & 1;
        int t  = (idx >> 8) % 18;
        int ck = idx / BCHUNK;
        int pr = ck * 16 + hb * 8 + j;   // K-pair row 0..559
        int g  = t * 16 + c;
        float d[2] = {0.f, 0.f};
        if (g < G_DIM) {
            for (int half = 0; half < 2; ++half) {
                int cc = 2 * pr + half;
                if (cc < C_DIM) {
                    float nn = 0.f;
                    for (int h = 0; h < H_DIM; ++h) { float w = ew[cc*H_DIM+h]; nn += w*w; }
                    nn = sqrtf(nn);
                    float sc = (nn > 1.0f) ? 1.0f / (nn + 1e-7f) : 1.0f;
                    float dd = 0.f;
                    for (int h = 0; h < H_DIM; ++h) dd += ew[cc*H_DIM+h] * wih[g*H_DIM+h];
                    d[half] = dd * sc;
                }
            }
        }
        wcP[idx] = pack2(d[0], d[1]);
    } else if (idx < WC_N + WHH_N) {
        int j = idx - WC_N;
        int kp = j / GP, g = j % GP;
        int k0 = 2 * kp, k1 = k0 + 1;
        float a = (g < G_DIM && k0 < H_DIM) ? whh[g*H_DIM+k0] : 0.f;
        float b = (g < G_DIM && k1 < H_DIM) ? whh[g*H_DIM+k1] : 0.f;
        whhP[j] = pack2(a, b);
    } else if (idx < WC_N + WHH_N + GP) {
        int j = idx - WC_N - WHH_N;
        bias[j] = (j < G_DIM) ? (bih[j] + bhh[j]) : 0.f;
    }
}

// ---------------------------------------------------------------------------
// K1: xg = x @ Wc + bias  (65536x1120x288, bf16 WMMA)
// Block: 512 thr (16 waves), tile M=128 x N=288, K-chunks of 32.
// Double-buffered LDS; async B copy for chunk k+1 overlaps WMMA of chunk k.
// Output written TILE-MAJOR for the LSTM: xg[btile][s][bl(16)][GP]
// ---------------------------------------------------------------------------
__global__ __launch_bounds__(512)
void gemm_xg_kernel(const float* __restrict__ x, const unsigned* __restrict__ wcP,
                    const float* __restrict__ bias, float* __restrict__ xg) {
    __shared__ __align__(16) unsigned Asm[2][128][16];   // 128 rows x 16 K-pairs
    __shared__ __align__(16) unsigned Bsm[2][BCHUNK];    // fragment-major chunk slab
    __shared__ float biasS[GP];

    const int tid  = threadIdx.x;
    const int lane = tid & 31;
    const int wave = tid >> 5;
    const int m    = wave >> 1;
    const int half = wave & 1;
    const int nbase = blockIdx.x * 128;

    if (tid < GP) biasS[tid] = bias[tid];

    FragF acc[9];
#pragma unroll
    for (int t = 0; t < 9; ++t)
#pragma unroll
        for (int v = 0; v < 8; ++v) acc[t].f[v] = 0.f;

    const int r   = tid >> 2;   // staged row 0..127
    const int seg = tid & 3;    // 8-float segment of the 32-K chunk

    auto issueB = [&](int ck, int pb) {
        const unsigned* gsrc = wcP + (size_t)ck * BCHUNK;
        unsigned ldsBase = ldsAddr(&Bsm[pb][0]);
#pragma unroll
        for (int i = 0; i < 3; ++i) {
            int q = tid + i * 512;
            if (q < 1152)
                asyncCopyB128(ldsBase + q * 16,
                              (unsigned long long)(uintptr_t)(gsrc + q * 4));
        }
    };
    auto stageA = [&](int ck, int pb) {
        int c0 = ck * 32 + seg * 8;
        const float* xr = x + (size_t)(nbase + r) * C_DIM + c0;
        float f[8];
        if (c0 + 8 <= C_DIM) {
            float4 a0 = *(const float4*)(xr);
            float4 a1 = *(const float4*)(xr + 4);
            f[0]=a0.x; f[1]=a0.y; f[2]=a0.z; f[3]=a0.w;
            f[4]=a1.x; f[5]=a1.y; f[6]=a1.z; f[7]=a1.w;
        } else {
#pragma unroll
            for (int q = 0; q < 8; ++q) f[q] = (c0 + q < C_DIM) ? xr[q] : 0.f;
        }
#pragma unroll
        for (int q = 0; q < 4; ++q) Asm[pb][r][seg*4 + q] = pack2(f[2*q], f[2*q+1]);
    };

    issueB(0, 0);
    stageA(0, 0);
    asyncWait0();
    __syncthreads();

    int pb = 0;
    for (int ck = 0; ck < KCH; ++ck) {
        if (ck + 1 < KCH) {             // prefetch next chunk into other buffer
            issueB(ck + 1, pb ^ 1);
            stageA(ck + 1, pb ^ 1);
        }
        // ---- A fragment (16-bit A 16x32 layout): 2x ds_load_b128 ---------
        FragH afr;
        {
            int rr = m * 16 + (lane & 15);
            int kb = (lane < 16) ? 0 : 8;
#pragma unroll
            for (int j = 0; j < 8; ++j) {
                int p = j + kb;
                afr.u[j] = Asm[pb][rr][p];
            }
        }
        // ---- B fragments (contiguous 8 dwords -> 2x ds_load_b128) + WMMA -
#pragma unroll
        for (int t = 0; t < 9; ++t) {
            int fb = (((half * 9 + t) * 2 + ((lane < 16) ? 0 : 1)) * 16 + (lane & 15)) * 8;
            FragH bfr;
#pragma unroll
            for (int j = 0; j < 8; ++j) bfr.u[j] = Bsm[pb][fb + j];
            acc[t].v = __builtin_amdgcn_wmma_f32_16x16x32_bf16(
                false, afr.v, false, bfr.v, (short)0, acc[t].v, false, false);
        }
        if (ck + 1 < KCH) asyncWait0();  // DMA ran concurrent with the WMMAs
        __syncthreads();
        pb ^= 1;
    }

    // ---- epilogue: + bias, store f32 in LSTM tile-major layout ----------
    const int b     = nbase >> 9;        // batch row (block never crosses b)
    const int sbase = nbase & 511;
    const int btile = b >> 4, bl = b & 15;
#pragma unroll
    for (int t = 0; t < 9; ++t) {
        int g = (half * 9 + t) * 16 + (lane & 15);
        float bv = biasS[g];
#pragma unroll
        for (int v = 0; v < 8; ++v) {
            int M = (lane < 16) ? v : v + 8;
            int s = sbase + m * 16 + M;
            xg[((size_t)(btile * 512 + s) * 16 + bl) * GP + g] = acc[t].f[v] + bv;
        }
    }
}

// ---------------------------------------------------------------------------
// K2: LSTM scan. 8 blocks x 288 thr (9 waves); block owns 16 batch rows.
// Per step: gates = xgTile + h @ w_hh^T via WMMA; xg tile for step st+1 is
// async-DMA'd into a double-buffered LDS slab while step st computes.
// ---------------------------------------------------------------------------
__global__ __launch_bounds__(288)
void lstm_kernel(const float* __restrict__ xg, const unsigned* __restrict__ whhP,
                 float* __restrict__ outb) {
    __shared__ float    hS[16][72];
    __shared__ __align__(16) unsigned Apk[16][48];   // h bf16 pairs, K pad 96
    __shared__ float    Gact[16][GP];
    __shared__ __align__(16) float Xg[2][16][GP];    // per-step gate tile

    const int tid   = threadIdx.x;
    const int lane  = tid & 31;
    const int wave  = tid >> 5;          // 0..8 -> tiles 2w, 2w+1
    const int btile = blockIdx.x;
    const int mbase = btile * 16;

    // loop-invariant B fragments (w_hh^T) -> registers
    unsigned bfr[2][3][8];
#pragma unroll
    for (int tt = 0; tt < 2; ++tt) {
        int col = (wave * 2 + tt) * 16 + (lane & 15);
        int pbr = (lane < 16) ? 0 : 8;
#pragma unroll
        for (int k = 0; k < 3; ++k)
#pragma unroll
            for (int j = 0; j < 8; ++j)
                bfr[tt][k][j] = whhP[(size_t)(k * 16 + pbr + j) * GP + col];
    }

    auto issueXg = [&](int st, int pb) {
        const float* gsrc = xg + (size_t)(btile * 512 + st) * 16 * GP; // 18 KB slab
        unsigned ldsBase = ldsAddr(&Xg[pb][0][0]);
#pragma unroll
        for (int i = 0; i < 4; ++i) {
            int q = tid + i * 288;
            if (q < 1152)
                asyncCopyB128(ldsBase + q * 16,
                              (unsigned long long)(uintptr_t)(gsrc + q * 4));
        }
    };

    issueXg(0, 0);
    for (int i = tid; i < 16 * 72; i += 288) ((float*)hS)[i] = 0.f;
    float cst[4] = {0.f, 0.f, 0.f, 0.f};
    asyncWait0();
    __syncthreads();

    int pb = 0;
    for (int st = 0; st < S_DIM; ++st) {
        if (st + 1 < S_DIM) issueXg(st + 1, pb ^ 1);   // overlaps whole step

        // pack h_prev -> bf16 A layout (zero pad K 70..95)
        for (int i = tid; i < 16 * 48; i += 288) {
            int rr = i / 48, pp = i % 48;
            int k0 = 2 * pp, k1 = k0 + 1;
            float a = (k0 < H_DIM) ? hS[rr][k0] : 0.f;
            float b = (k1 < H_DIM) ? hS[rr][k1] : 0.f;
            ((unsigned*)Apk)[i] = pack2(a, b);
        }
        __syncthreads();

        FragH afr[3];
        {
            int rr = lane & 15;
            int kb = (lane < 16) ? 0 : 8;
#pragma unroll
            for (int k = 0; k < 3; ++k)
#pragma unroll
                for (int j = 0; j < 8; ++j) {
                    int p = j + kb;
                    afr[k].u[j] = Apk[rr][k * 16 + p];
                }
        }
#pragma unroll
        for (int tt = 0; tt < 2; ++tt) {
            int g = (wave * 2 + tt) * 16 + (lane & 15);
            FragF acc;
#pragma unroll
            for (int v = 0; v < 8; ++v) {
                int M = (lane < 16) ? v : v + 8;
                acc.f[v] = Xg[pb][M][g];
            }
#pragma unroll
            for (int k = 0; k < 3; ++k) {
                FragH bb;
#pragma unroll
                for (int j = 0; j < 8; ++j) bb.u[j] = bfr[tt][k][j];
                acc.v = __builtin_amdgcn_wmma_f32_16x16x32_bf16(
                    false, afr[k].v, false, bb.v, (short)0, acc.v, false, false);
            }
#pragma unroll
            for (int v = 0; v < 8; ++v) {
                int M = (lane < 16) ? v : v + 8;
                Gact[M][g] = acc.f[v];
            }
        }
        __syncthreads();

        // elementwise gate math: i,f,g,o at column offsets 0/70/140/210
#pragma unroll
        for (int i = 0; i < 4; ++i) {
            int p = tid + i * 288;
            if (p < 16 * H_DIM) {
                int rr = p / H_DIM, h = p % H_DIM;
                float pi = Gact[rr][h];
                float pf = Gact[rr][H_DIM + h];
                float pg = Gact[rr][2 * H_DIM + h];
                float po = Gact[rr][3 * H_DIM + h];
                float ig = 1.f / (1.f + __expf(-pi));
                float fg = 1.f / (1.f + __expf(-pf));
                float gg = tanhf(pg);
                float og = 1.f / (1.f + __expf(-po));
                float cn = fg * cst[i] + ig * gg;
                cst[i] = cn;
                float hv = og * tanhf(cn);
                hS[rr][h] = hv;
                outb[((size_t)(mbase + rr) * S_DIM + st) * OSTR + h] = hv;
            }
        }
        if (st + 1 < S_DIM) asyncWait0();   // next-step tile DMA done
        __syncthreads();
        pb ^= 1;
    }
}

// ---------------------------------------------------------------------------
// K3: en[n] = relu(out[n] @ w1^T + b1) @ w2^T + b2
// ---------------------------------------------------------------------------
__global__ __launch_bounds__(256)
void attn_energy_kernel(const float* __restrict__ outb, const float* __restrict__ w1,
                        const float* __restrict__ b1, const float* __restrict__ w2,
                        const float* __restrict__ b2, float* __restrict__ en) {
    __shared__ float w1S[64][72];
    __shared__ float b1S[64], w2S[64];
    int tid = threadIdx.x;
    for (int i = tid; i < 64 * H_DIM; i += 256) { w1S[i / H_DIM][i % H_DIM] = w1[i]; }
    if (tid < 64) { b1S[tid] = b1[tid]; w2S[tid] = w2[tid]; }
    __syncthreads();
    int n = blockIdx.x * 256 + tid;
    const float* orow = outb + (size_t)n * OSTR;
    float o[H_DIM];
#pragma unroll
    for (int h = 0; h < H_DIM; ++h) o[h] = orow[h];
    float e = b2[0];
    for (int j = 0; j < 64; ++j) {
        float s = b1S[j];
#pragma unroll
        for (int h = 0; h < H_DIM; ++h) s += o[h] * w1S[j][h];
        e += fmaxf(s, 0.f) * w2S[j];
    }
    en[n] = e;
}

// ---------------------------------------------------------------------------
// K4: per-batch softmax over S, attention pool, FC(3), softmax -> d_out
// ---------------------------------------------------------------------------
__global__ __launch_bounds__(256)
void pool_fc_kernel(const float* __restrict__ outb, const float* __restrict__ en,
                    const float* __restrict__ fcw, const float* __restrict__ fcb,
                    float* __restrict__ dout) {
    __shared__ float red[256];
    __shared__ float wts[512];
    __shared__ float pooled[72];
    int b = blockIdx.x, tid = threadIdx.x;
    float e0 = en[b * 512 + tid];
    float e1 = en[b * 512 + 256 + tid];
    red[tid] = fmaxf(e0, e1);
    __syncthreads();
    for (int s = 128; s > 0; s >>= 1) {
        if (tid < s) red[tid] = fmaxf(red[tid], red[tid + s]);
        __syncthreads();
    }
    float mx = red[0];
    __syncthreads();
    float x0 = __expf(e0 - mx), x1 = __expf(e1 - mx);
    red[tid] = x0 + x1;
    __syncthreads();
    for (int s = 128; s > 0; s >>= 1) {
        if (tid < s) red[tid] += red[tid + s];
        __syncthreads();
    }
    float inv = 1.f / red[0];
    wts[tid] = x0 * inv;
    wts[tid + 256] = x1 * inv;
    __syncthreads();
    if (tid < H_DIM) {
        float acc = 0.f;
        for (int s = 0; s < 512; ++s)
            acc += wts[s] * outb[(size_t)(b * 512 + s) * OSTR + tid];
        pooled[tid] = acc;
    }
    __syncthreads();
    if (tid == 0) {
        float lg[3]; float mmax = -1e30f;
        for (int k = 0; k < 3; ++k) {
            float s = fcb[k];
            for (int h = 0; h < H_DIM; ++h) s += fcw[k * H_DIM + h] * pooled[h];
            lg[k] = s; mmax = fmaxf(mmax, s);
        }
        float sum = 0.f;
        for (int k = 0; k < 3; ++k) { lg[k] = __expf(lg[k] - mmax); sum += lg[k]; }
        for (int k = 0; k < 3; ++k) dout[b * 3 + k] = lg[k] / sum;
    }
}

// ---------------------------------------------------------------------------
extern "C" void kernel_launch(void* const* d_in, const int* in_sizes, int n_in,
                              void* d_out, int out_size, void* d_ws, size_t ws_size,
                              hipStream_t stream) {
    (void)in_sizes; (void)n_in; (void)out_size; (void)ws_size;
    const float* x   = (const float*)d_in[0];
    const float* ew  = (const float*)d_in[1];
    const float* wih = (const float*)d_in[2];
    const float* whh = (const float*)d_in[3];
    const float* bih = (const float*)d_in[4];
    const float* bhh = (const float*)d_in[5];
    const float* aw1 = (const float*)d_in[6];
    const float* ab1 = (const float*)d_in[7];
    const float* aw2 = (const float*)d_in[8];
    const float* ab2 = (const float*)d_in[9];
    const float* fcw = (const float*)d_in[10];
    const float* fcb = (const float*)d_in[11];
    float* dout = (float*)d_out;

    char* ws = (char*)d_ws;
    size_t off = 0;
    auto alloc = [&](size_t bytes) {
        size_t o = off; off = (off + bytes + 255) & ~(size_t)255; return o;
    };
    unsigned* wcP  = (unsigned*)(ws + alloc((size_t)KCH * BCHUNK * 4));  // 645 KB
    unsigned* whhP = (unsigned*)(ws + alloc(48ull * GP * 4));            // 55 KB
    float*    bias = (float*)   (ws + alloc(GP * 4));
    float*    xg   = (float*)   (ws + alloc((size_t)NROWS * GP * 4));    // 75.5 MB
    float*    outb = (float*)   (ws + alloc((size_t)NROWS * OSTR * 4));  // 18.9 MB
    float*    en   = (float*)   (ws + alloc((size_t)NROWS * 4));         // 256 KB

    prep_kernel<<<686, 256, 0, stream>>>(ew, wih, whh, bih, bhh, wcP, whhP, bias);
    gemm_xg_kernel<<<512, 512, 0, stream>>>(x, wcP, bias, xg);
    lstm_kernel<<<8, 288, 0, stream>>>(xg, whhP, outb);
    attn_energy_kernel<<<256, 256, 0, stream>>>(outb, aw1, ab1, aw2, ab2, en);
    pool_fc_kernel<<<128, 256, 0, stream>>>(outb, en, fcw, fcb, dout);
}
